// litho_23021024707062
// MI455X (gfx1250) — matrix operand order repaired and checked by reading, not compiled, not run on hardware
//
#include <hip/hip_runtime.h>

typedef __attribute__((ext_vector_type(16))) _Float16 v16h;
typedef __attribute__((ext_vector_type(8)))  float    v8f;

#define HS   1024
#define NCH  24
#define KDIM 35
#define KPAD 96                      // 2*35 = 70 interleaved cos/sin cols, padded to 3*32
#define TWO_PI_OVER_N 0.006135923151542565f

// ---- mask = sigmoid(4*(avgpool3(mask_s) - 0.5)) -----------------------------
__global__ void pool_mask_kernel(const float* __restrict__ ms, float* __restrict__ mask) {
  int idx = blockIdx.x * blockDim.x + threadIdx.x;
  int y = idx >> 10, x = idx & 1023;
  float s = 0.f;
  for (int dy = -1; dy <= 1; ++dy) {
    int yy = y + dy; if (yy < 0 || yy >= HS) continue;
    for (int dx = -1; dx <= 1; ++dx) {
      int xx = x + dx; if (xx < 0 || xx >= HS) continue;
      s += ms[yy * HS + xx];
    }
  }
  float p = s * (1.0f / 9.0f);
  mask[y * HS + x] = 1.0f / (1.0f + __expf(-4.0f * (p - 0.5f)));
}

// ---- partial DFT along y: S[u][x] = sum_y mask[y][x] e^{-2pi i u_orig y/N} --
__global__ void dfty_kernel(const float* __restrict__ mask, float* __restrict__ S) {
  int u = blockIdx.y;
  int x = blockIdx.x * blockDim.x + threadIdx.x;
  int uo = (1007 + u) & 1023;                       // unshifted freq index (OFF=495)
  float sre = 0.f, sim = 0.f;
  for (int y = 0; y < HS; ++y) {
    float m = mask[y * HS + x];
    float a = (float)((uo * y) & 1023) * TWO_PI_OVER_N;   // exact int phase, no drift
    float c, sn; __sincosf(a, &sn, &c);
    sre += m * c; sim -= m * sn;
  }
  S[(u * HS + x) * 2 + 0] = sre;
  S[(u * HS + x) * 2 + 1] = sim;
}

// ---- crop[u][v] = sum_x S[u][x] e^{-2pi i v_orig x/N} -----------------------
__global__ void dftx_kernel(const float* __restrict__ S, float* __restrict__ crop) {
  int u = blockIdx.x, v = threadIdx.x;
  if (v >= KDIM) return;
  int vo = (1007 + v) & 1023;
  float cr = 0.f, ci = 0.f;
  for (int x = 0; x < HS; ++x) {
    float sr = S[(u * HS + x) * 2 + 0];
    float si = S[(u * HS + x) * 2 + 1];
    float a = (float)((vo * x) & 1023) * TWO_PI_OVER_N;
    float c, sn; __sincosf(a, &sn, &c);
    cr += sr * c + si * sn;            // (sr+i si)(cos - i sin)
    ci += si * c - sr * sn;
  }
  crop[(u * KDIM + v) * 2 + 0] = cr;
  crop[(u * KDIM + v) * 2 + 1] = ci;
}

// ---- A matrix, pre-swizzled into per-lane WMMA fragment order ---------------
// A[y][2i]=cos(2pi i y/N), A[y][2i+1]=sin(...). A 16x32 lane order used here (B uses the same):
// lane L: row = L%16, half = L/16, element e -> K_local = half*16 + e.
__global__ void build_a_kernel(_Float16* __restrict__ A) {
  int ytile = blockIdx.x, s = blockIdx.y, lane = threadIdx.x;
  int row  = ytile * 16 + (lane & 15);
  int half = lane >> 4;
  _Float16* dst = A + ((size_t)(ytile * 3 + s) * 32 + lane) * 16;
  for (int e = 0; e < 16; ++e) {
    int kk = s * 32 + (half << 4) + e;
    float v = 0.f;
    if (kk < 2 * KDIM) {
      int i = kk >> 1;
      float a = (float)((i * row) & 1023) * TWO_PI_OVER_N;
      v = (kk & 1) ? __sinf(a) : __cosf(a);
    }
    dst[e] = (_Float16)v;
  }
}

// ---- per (branch, channel): T[i][x] = sum_j crop*ker*dose/N^2 * e^{+2pi i jx/N}
// stored as K-contiguous f16 panels: Bre[k][x][2i]=Tre, [2i+1]=-Tim ; Bim: Tim, Tre
__global__ void build_b_kernel(const float* __restrict__ crop,
                               const float* __restrict__ kre, const float* __restrict__ kim,
                               float dose, _Float16* __restrict__ Bre, _Float16* __restrict__ Bim) {
  int x = blockIdx.x * blockDim.x + threadIdx.x;
  int i = blockIdx.y, k = blockIdx.z;
  const float inv = dose * (1.0f / 1048576.0f);    // fold ifft 1/N^2 early -> f16 safe
  float tre = 0.f, tim = 0.f;
  for (int j = 0; j < KDIM; ++j) {
    float pr = crop[(i * KDIM + j) * 2 + 0], pi = crop[(i * KDIM + j) * 2 + 1];
    float kr = kre[(k * KDIM + i) * KDIM + j], ki = kim[(k * KDIM + i) * KDIM + j];
    float cr = (pr * kr - pi * ki) * inv;
    float ci = (pr * ki + pi * kr) * inv;
    float a = (float)((j * x) & 1023) * TWO_PI_OVER_N;
    float c, sn; __sincosf(a, &sn, &c);
    tre += cr * c - ci * sn;                        // (cr+i ci)(cos + i sin)
    tim += cr * sn + ci * c;
  }
  _Float16* br = Bre + (size_t)(k * HS + x) * KPAD;
  _Float16* bi = Bim + (size_t)(k * HS + x) * KPAD;
  br[2 * i] = (_Float16)tre;  br[2 * i + 1] = (_Float16)(-tim);
  bi[2 * i] = (_Float16)tim;  bi[2 * i + 1] = (_Float16)tre;
  if (i == 0)
    for (int c2 = 2 * KDIM; c2 < KPAD; ++c2) { br[c2] = (_Float16)0.f; bi[c2] = (_Float16)0.f; }
}

// ---- big matmul: inten = sum_k scale_k*((A@Bre_k)^2 + (A@Bim_k)^2); sigmoid --
__global__ void __launch_bounds__(256) wmma_litho_kernel(
    const _Float16* __restrict__ A, const _Float16* __restrict__ Bre,
    const _Float16* __restrict__ Bim, const float* __restrict__ scale,
    float* __restrict__ out) {
  int lane  = threadIdx.x & 31;
  int wave  = threadIdx.x >> 5;
  int ytile = blockIdx.y * 8 + wave;     // 8 waves share x-columns -> L1/L2 reuse of B
  int xtile = blockIdx.x;
  int half  = lane >> 4;
  int nloc  = lane & 15;
  int x     = xtile * 16 + nloc;

  const v16h* abase = (const v16h*)A;
  v16h a0 = abase[(size_t)(ytile * 3 + 0) * 32 + lane];
  v16h a1 = abase[(size_t)(ytile * 3 + 1) * 32 + lane];
  v16h a2 = abase[(size_t)(ytile * 3 + 2) * 32 + lane];

  v8f inten = {0, 0, 0, 0, 0, 0, 0, 0};
  for (int k = 0; k < NCH; ++k) {
    const _Float16* brp = Bre + (size_t)(k * HS + x) * KPAD + half * 16;
    const _Float16* bip = Bim + (size_t)(k * HS + x) * KPAD + half * 16;
    if (k + 1 < NCH) {                   // stage next channel into cache
      __builtin_prefetch(brp + (size_t)HS * KPAD, 0, 1);
      __builtin_prefetch(bip + (size_t)HS * KPAD, 0, 1);
    }
    v16h br0 = *(const v16h*)(brp), br1 = *(const v16h*)(brp + 32), br2 = *(const v16h*)(brp + 64);
    v16h bi0 = *(const v16h*)(bip), bi1 = *(const v16h*)(bip + 32), bi2 = *(const v16h*)(bip + 64);
    v8f ar = {0, 0, 0, 0, 0, 0, 0, 0};
    v8f ai = {0, 0, 0, 0, 0, 0, 0, 0};
    ar = __builtin_amdgcn_wmma_f32_16x16x32_f16(false, a0, false, br0, (short)0, ar, false, false);
    ar = __builtin_amdgcn_wmma_f32_16x16x32_f16(false, a1, false, br1, (short)0, ar, false, false);
    ar = __builtin_amdgcn_wmma_f32_16x16x32_f16(false, a2, false, br2, (short)0, ar, false, false);
    ai = __builtin_amdgcn_wmma_f32_16x16x32_f16(false, a0, false, bi0, (short)0, ai, false, false);
    ai = __builtin_amdgcn_wmma_f32_16x16x32_f16(false, a1, false, bi1, (short)0, ai, false, false);
    ai = __builtin_amdgcn_wmma_f32_16x16x32_f16(false, a2, false, bi2, (short)0, ai, false, false);
    float sc = scale[k];
    #pragma unroll
    for (int r = 0; r < 8; ++r) inten[r] += sc * (ar[r] * ar[r] + ai[r] * ai[r]);
  }
  #pragma unroll
  for (int r = 0; r < 8; ++r) {          // C/D layout: M = half*8 + r, N = lane%16
    int row = ytile * 16 + half * 8 + r;
    out[row * HS + xtile * 16 + nloc] =
        1.0f / (1.0f + __expf(-50.0f * (inten[r] - 0.225f)));
  }
}

extern "C" void kernel_launch(void* const* d_in, const int* in_sizes, int n_in,
                              void* d_out, int out_size, void* d_ws, size_t ws_size,
                              hipStream_t stream) {
  const float* mask_s = (const float*)d_in[0];
  const float* kf_re  = (const float*)d_in[1];
  const float* kf_im  = (const float*)d_in[2];
  const float* kd_re  = (const float*)d_in[3];
  const float* kd_im  = (const float*)d_in[4];
  const float* kf_sc  = (const float*)d_in[5];
  const float* kd_sc  = (const float*)d_in[6];
  float* out = (float*)d_out;

  char* ws = (char*)d_ws;                       // ~14.1 MB total, 2KB-aligned slices
  float*    mask = (float*)(ws + 0x000000);     // 4 MB
  float*    S    = (float*)(ws + 0x400000);     // 35*1024*2 f32
  float*    crop = (float*)(ws + 0x446000);     // 35*35*2 f32
  _Float16* A    = (_Float16*)(ws + 0x448800);  // 64*3*32*16 f16 (pre-swizzled)
  _Float16* Bre  = (_Float16*)(ws + 0x478800);  // 24*1024*96 f16
  _Float16* Bim  = (_Float16*)(ws + 0x8F8800);  // 24*1024*96 f16

  pool_mask_kernel<<<4096, 256, 0, stream>>>(mask_s, mask);
  build_a_kernel<<<dim3(64, 3), 32, 0, stream>>>(A);
  dfty_kernel<<<dim3(4, KDIM), 256, 0, stream>>>(mask, S);
  dftx_kernel<<<KDIM, 64, 0, stream>>>(S, crop);

  const float* bre[3]  = {kf_re, kf_re, kd_re};
  const float* bim[3]  = {kf_im, kf_im, kd_im};
  const float* bsc[3]  = {kf_sc, kf_sc, kd_sc};
  const float  dose[3] = {1.0f, 1.02f, 0.98f};
  for (int b = 0; b < 3; ++b) {
    build_b_kernel<<<dim3(4, KDIM, NCH), 256, 0, stream>>>(crop, bre[b], bim[b], dose[b], Bre, Bim);
    wmma_litho_kernel<<<dim3(64, 8), 256, 0, stream>>>(A, Bre, Bim, bsc[b],
                                                       out + (size_t)b * HS * HS);
  }
}
